// MoE_54107997995489
// MI455X (gfx1250) — matrix operand order repaired and checked by reading, not compiled, not run on hardware
//
#include <hip/hip_runtime.h>
#include <cstdint>
#include <cstddef>

typedef __bf16 bf16;
typedef __attribute__((ext_vector_type(16))) __bf16 v16bf;
typedef __attribute__((ext_vector_type(8)))  float  v8f;
typedef int v4i __attribute__((__vector_size__(16)));

#define AS1 __attribute__((address_space(1)))
#define AS3 __attribute__((address_space(3)))

#if __has_builtin(__builtin_amdgcn_global_load_async_to_lds_b128) && \
    __has_builtin(__builtin_amdgcn_s_wait_asynccnt)
#define HAVE_ASYNC 1
#else
#define HAVE_ASYNC 0
#endif

namespace {

constexpr int TOK  = 8 * 2048;   // B*T tokens
constexpr int DIMD = 512;        // D
constexpr int RH   = 1024;       // router hidden
constexpr int HH   = 2048;       // expert hidden
constexpr int NE   = 8;          // experts
constexpr int TK   = 4;          // top-k

constexpr int BM = 128, BN = 128, BK = 32;
constexpr int TSTR = BK + 8;     // 40 elems = 80 B row stride (multiple of 16 B)

union FragBF { v16bf v; uint4 q[2]; };

// ---------------- elementwise utility kernels ----------------

__global__ void k_cvt(const float* __restrict__ in, bf16* __restrict__ out, size_t n) {
  size_t i  = (size_t)blockIdx.x * blockDim.x + threadIdx.x;
  size_t st = (size_t)gridDim.x * blockDim.x;
  for (; i < n; i += st) out[i] = (bf16)in[i];
}

__global__ void k_zero(float* __restrict__ p, size_t n) {
  size_t i  = (size_t)blockIdx.x * blockDim.x + threadIdx.x;
  size_t st = (size_t)gridDim.x * blockDim.x;
  for (; i < n; i += st) p[i] = 0.0f;
}

// ---------------- bf16 WMMA GEMM, 128x128x32 tile, 8 waves ----------------
// EPI 0: outB = bf16(acc + bias)
// EPI 1: outB = bf16( silu(acc + bias) * other )   (other may alias outB)
// EPI 2: outF += (acc + bias) * mask[m*NE+expert] * mscale
template<int EPI>
__global__ __launch_bounds__(256)
void k_gemm(const bf16* __restrict__ A, const bf16* __restrict__ W,
            const float* __restrict__ bias, const bf16* other,
            const float* __restrict__ mask, int expert, float mscale,
            bf16* outB, float* outF, int M, int N, int K)
{
#if HAVE_ASYNC
  __shared__ uint4 AsQ[2][BM * TSTR / 8];
  __shared__ uint4 BsQ[2][BN * TSTR / 8];
#else
  __shared__ uint4 AsQ[1][BM * TSTR / 8];
  __shared__ uint4 BsQ[1][BN * TSTR / 8];
#endif

  const int tid  = threadIdx.x;
  const int lane = tid & 31;
  const int wv   = tid >> 5;      // 0..7
  const int wm   = wv >> 1;       // 0..3 : 32-row group
  const int wn   = wv & 1;        // 0..1 : 64-col group
  const int m0   = blockIdx.y * BM;
  const int n0   = blockIdx.x * BN;
  const int lr   = lane & 15;
  const int kh   = lane >> 4;     // lane half

  const v8f vz = {0.f,0.f,0.f,0.f,0.f,0.f,0.f,0.f};
  v8f acc[2][4];
  #pragma unroll
  for (int i = 0; i < 2; ++i)
    #pragma unroll
    for (int j = 0; j < 4; ++j) acc[i][j] = vz;

  // ---- per-thread tile-fill coordinates (also used by the prefetch path) ----
  const int arow = tid >> 2;
  const int acc0 = (tid & 3) << 3;
  const int bk0  = tid >> 4;
  const int bn0c = (tid & 15) << 3;

  // compute-phase fragment loads + 8 WMMAs; depends only on the LDS buffer base
  auto compute = [&](const bf16* As, const bf16* Bs) {
    FragBF a[2], b[4];
    #pragma unroll
    for (int i = 0; i < 2; ++i) {
      const uint4* p = (const uint4*)(As + (wm * 32 + i * 16 + lr) * TSTR);
      a[i].q[0] = p[kh];        // lane lo: K0..7   | lane hi: K8..15
      a[i].q[1] = p[2 + kh];    // lane lo: K16..23 | lane hi: K24..31
    }
    #pragma unroll
    for (int j = 0; j < 4; ++j) {
      const uint4* p = (const uint4*)(Bs + (wn * 64 + j * 16 + lr) * TSTR);
      b[j].q[0] = p[kh];            // lane lo: K0..7   | lane hi: K8..15
      b[j].q[1] = p[2 + kh];        // lane lo: K16..23 | lane hi: K24..31
    }
    #pragma unroll
    for (int i = 0; i < 2; ++i)
      #pragma unroll
      for (int j = 0; j < 4; ++j)
        acc[i][j] = __builtin_amdgcn_wmma_f32_16x16x32_bf16(
            false, a[i].v, false, b[j].v, (short)0, acc[i][j], false, false);
  };

#if HAVE_ASYNC
  // ---------- async double-buffered pipeline ----------
  auto issueA = [&](int k0, int buf) {   // DMA: global row-major panel -> LDS (ASYNCcnt)
    #pragma unroll
    for (int it = 0; it < 2; ++it) {
      int c   = tid + it * 256;
      int row = c >> 2;
      int cc  = (c & 3) << 3;
      const bf16* src = A + (size_t)(m0 + row) * K + (k0 + cc);
      bf16*       dst = (bf16*)AsQ[buf] + row * TSTR + cc;
      __builtin_amdgcn_global_load_async_to_lds_b128(
          (AS1 v4i*)src, (AS3 v4i*)dst, 0, 0);
    }
  };
  auto loadB = [&](int k0, uint4 bv[2]) {
    #pragma unroll
    for (int it = 0; it < 2; ++it) {
      int c  = tid + it * 256;
      int kk = c >> 4;
      int nn = (c & 15) << 3;
      bv[it] = *(const uint4*)(W + (size_t)(k0 + kk) * N + (n0 + nn));
    }
  };
  auto storeB = [&](const uint4 bv[2], int buf) {   // transpose: [BK x BN] -> [BN][BK]
    unsigned short* BsS = (unsigned short*)BsQ[buf];
    #pragma unroll
    for (int it = 0; it < 2; ++it) {
      int c  = tid + it * 256;
      int kk = c >> 4;
      int nn = (c & 15) << 3;
      unsigned int ww[4] = {bv[it].x, bv[it].y, bv[it].z, bv[it].w};
      #pragma unroll
      for (int j = 0; j < 8; ++j)
        BsS[(nn + j) * TSTR + kk] = (unsigned short)(ww[j >> 1] >> ((j & 1) * 16));
    }
  };

  const int T = K / BK;
  uint4 bv[2];
  issueA(0, 0);
  loadB(0, bv);
  storeB(bv, 0);
  __builtin_amdgcn_s_wait_asynccnt(0);
  __syncthreads();

  for (int t = 0; t < T; ++t) {
    const int cur = t & 1;
    const int nxt = cur ^ 1;
    if (t + 1 < T) {
      issueA((t + 1) * BK, nxt);   // DMA next A panel while we do math
      loadB((t + 1) * BK, bv);
      storeB(bv, nxt);
    }
    compute((const bf16*)AsQ[cur], (const bf16*)BsQ[cur]);
    if (t + 1 < T) {
      __builtin_amdgcn_s_wait_asynccnt(0);
      __syncthreads();
    }
  }
#else
  // ---------- fallback: proven single-buffer loop ----------
  bf16* As = (bf16*)AsQ[0];
  unsigned short* BsS = (unsigned short*)BsQ[0];
  for (int k0 = 0; k0 < K; k0 += BK) {
    #pragma unroll
    for (int it = 0; it < 2; ++it) {
      int c   = tid + it * 256;
      int row = c >> 2;
      int cc  = (c & 3) << 3;
      uint4 v = *(const uint4*)(A + (size_t)(m0 + row) * K + (k0 + cc));
      *(uint4*)(As + row * TSTR + cc) = v;
    }
    #pragma unroll
    for (int it = 0; it < 2; ++it) {
      int c  = tid + it * 256;
      int kk = c >> 4;
      int nn = (c & 15) << 3;
      uint4 v = *(const uint4*)(W + (size_t)(k0 + kk) * N + (n0 + nn));
      unsigned int ww[4] = {v.x, v.y, v.z, v.w};
      #pragma unroll
      for (int j = 0; j < 8; ++j)
        BsS[(nn + j) * TSTR + kk] = (unsigned short)(ww[j >> 1] >> ((j & 1) * 16));
    }
    if (k0 + BK < K) {
      __builtin_prefetch(A + (size_t)(m0 + arow) * K + (k0 + BK) + acc0, 0, 0);
      __builtin_prefetch(W + (size_t)(k0 + BK + bk0) * N + n0 + bn0c, 0, 0);
    }
    __syncthreads();
    compute(As, (const bf16*)BsQ[0]);
    __syncthreads();
  }
#endif

  // Epilogue (C/D layout: VGPR r -> M = r + 8*laneHalf, N = lane&15)
  #pragma unroll
  for (int i = 0; i < 2; ++i) {
    #pragma unroll
    for (int j = 0; j < 4; ++j) {
      const int n    = n0 + wn * 64 + j * 16 + lr;
      const float bn = bias[n];
      const int mb   = m0 + wm * 32 + i * 16 + kh * 8;
      #pragma unroll
      for (int r = 0; r < 8; ++r) {
        const int m = mb + r;
        const size_t idx = (size_t)m * N + n;
        float v = acc[i][j][r] + bn;
        if (EPI == 0) {
          outB[idx] = (bf16)v;
        } else if (EPI == 1) {
          float s = v / (1.0f + __expf(-v));        // silu
          outB[idx] = (bf16)(s * (float)other[idx]);
        } else {
          float wgt = mask[(size_t)m * NE + expert] * mscale;
          outF[idx] += v * wgt;
        }
      }
    }
  }
}

// ---------------- router logits + top-k mask (one wave per token) ----------------
__global__ __launch_bounds__(256)
void k_router_mask(const bf16* __restrict__ rh, const bf16* __restrict__ w,
                   const float* __restrict__ bias, float* __restrict__ mask,
                   int M, int K)
{
  const int gw   = (int)(((size_t)blockIdx.x * blockDim.x + threadIdx.x) >> 5);
  const int lane = threadIdx.x & 31;
  if (gw >= M) return;
  const bf16* row = rh + (size_t)gw * K;
  float acc[NE];
  #pragma unroll
  for (int e = 0; e < NE; ++e) acc[e] = 0.f;
  for (int k = lane; k < K; k += 32) {
    float xv = (float)row[k];
    uint4 q = *(const uint4*)(w + (size_t)k * NE);   // 8 bf16 weights for this k
    unsigned int ww[4] = {q.x, q.y, q.z, q.w};
    #pragma unroll
    for (int e = 0; e < NE; ++e) {
      unsigned int word = ww[e >> 1];
      unsigned int bits = (e & 1) ? (word & 0xffff0000u) : (word << 16);
      acc[e] += xv * __uint_as_float(bits);
    }
  }
  #pragma unroll
  for (int e = 0; e < NE; ++e) {
    float s = acc[e];
    #pragma unroll
    for (int off = 16; off > 0; off >>= 1) s += __shfl_down(s, off, 32);
    acc[e] = s;
  }
  if (lane == 0) {
    float lg[NE];
    #pragma unroll
    for (int e = 0; e < NE; ++e) lg[e] = acc[e] + bias[e];
    #pragma unroll
    for (int e = 0; e < NE; ++e) {
      int rank = 0;
      #pragma unroll
      for (int j = 0; j < NE; ++j)
        rank += (lg[j] > lg[e]) || ((lg[j] == lg[e]) && (j < e));   // top_k tie-break
      mask[(size_t)gw * NE + e] = (rank < TK) ? 1.0f : 0.0f;
    }
  }
}

} // namespace

extern "C" void kernel_launch(void* const* d_in, const int* in_sizes, int n_in,
                              void* d_out, int out_size, void* d_ws, size_t ws_size,
                              hipStream_t stream)
{
  (void)in_sizes; (void)n_in; (void)ws_size;
  const float* x_f     = (const float*)d_in[0];
  const float* r_in_w  = (const float*)d_in[1];
  const float* r_in_b  = (const float*)d_in[2];
  const float* rs_w1   = (const float*)d_in[3];
  const float* rs_b1   = (const float*)d_in[4];
  const float* rs_w2   = (const float*)d_in[5];
  const float* rs_b2   = (const float*)d_in[6];
  const float* rs_w3   = (const float*)d_in[7];
  const float* rs_b3   = (const float*)d_in[8];
  const float* r_out_w = (const float*)d_in[9];
  const float* r_out_b = (const float*)d_in[10];
  const float* e_in_w  = (const float*)d_in[11];
  const float* e_in_b  = (const float*)d_in[12];
  const float* es_w1   = (const float*)d_in[13];
  const float* es_b1   = (const float*)d_in[14];
  const float* es_w2   = (const float*)d_in[15];
  const float* es_b2   = (const float*)d_in[16];
  const float* es_w3   = (const float*)d_in[17];
  const float* es_b3   = (const float*)d_in[18];
  const float* e_out_w = (const float*)d_in[19];
  const float* e_out_b = (const float*)d_in[20];
  float* out = (float*)d_out;

  char* ws = (char*)d_ws;
  size_t off = 0;
  auto carve = [&](size_t elems, size_t esz) -> void* {
    void* p = ws + off;
    off += (elems * esz + 255) & ~(size_t)255;
    return p;
  };
  bf16* x_bf   = (bf16*)carve((size_t)TOK * DIMD, 2);
  bf16* rw_in  = (bf16*)carve((size_t)DIMD * RH, 2);
  bf16* rw1    = (bf16*)carve((size_t)RH * RH, 2);
  bf16* rw2    = (bf16*)carve((size_t)RH * RH, 2);
  bf16* rw3    = (bf16*)carve((size_t)RH * RH, 2);
  bf16* rw_out = (bf16*)carve((size_t)RH * NE, 2);
  float* maskp = (float*)carve((size_t)TOK * NE, 4);
  bf16* buf0   = (bf16*)carve((size_t)TOK * HH, 2);
  bf16* buf1   = (bf16*)carve((size_t)TOK * HH, 2);
  bf16* ew_in  = (bf16*)carve((size_t)DIMD * HH, 2);
  bf16* ew1    = (bf16*)carve((size_t)HH * HH, 2);
  bf16* ew2    = (bf16*)carve((size_t)HH * HH, 2);
  bf16* ew3    = (bf16*)carve((size_t)HH * HH, 2);
  bf16* ew_out = (bf16*)carve((size_t)HH * DIMD, 2);

  auto cvt = [&](const float* s, bf16* d, size_t n) {
    int blocks = (int)((n + 2047) / 2048);
    if (blocks > 4096) blocks = 4096;
    k_cvt<<<blocks, 256, 0, stream>>>(s, d, n);
  };
  auto gemm0 = [&](const bf16* A, const bf16* W, const float* b, bf16* o,
                   int M, int N, int K) {
    dim3 g(N / BN, M / BM);
    k_gemm<0><<<g, 256, 0, stream>>>(A, W, b, nullptr, nullptr, 0, 0.f, o, nullptr, M, N, K);
  };
  auto gemm1 = [&](const bf16* A, const bf16* W, const float* b, const bf16* oth,
                   bf16* o, int M, int N, int K) {
    dim3 g(N / BN, M / BM);
    k_gemm<1><<<g, 256, 0, stream>>>(A, W, b, oth, nullptr, 0, 0.f, o, nullptr, M, N, K);
  };
  auto gemm2 = [&](const bf16* A, const bf16* W, const float* b, int e, float sc,
                   float* o, int M, int N, int K) {
    dim3 g(N / BN, M / BM);
    k_gemm<2><<<g, 256, 0, stream>>>(A, W, b, nullptr, maskp, e, sc, nullptr, o, M, N, K);
  };

  // zero output accumulator (d_ws/d_out are poisoned by the harness)
  k_zero<<<4096, 256, 0, stream>>>(out, (size_t)out_size);

  // bf16 conversions: activations + router weights (once)
  cvt(x_f,     x_bf,  (size_t)TOK * DIMD);
  cvt(r_in_w,  rw_in, (size_t)DIMD * RH);
  cvt(rs_w1,   rw1,   (size_t)RH * RH);
  cvt(rs_w2,   rw2,   (size_t)RH * RH);
  cvt(rs_w3,   rw3,   (size_t)RH * RH);
  cvt(r_out_w, rw_out,(size_t)RH * NE);

  // ---- router ----
  gemm0(x_bf, rw_in, r_in_b, buf0, TOK, RH, DIMD);      // rh   = x @ Wr_in + b
  gemm0(buf0, rw2,   rs_b2,  buf1, TOK, RH, RH);        // h2   = rh @ w2 + b2
  gemm1(buf0, rw1,   rs_b1,  buf1, buf1, TOK, RH, RH);  // hs   = silu(rh@w1+b1) * h2
  gemm0(buf1, rw3,   rs_b3,  buf0, TOK, RH, RH);        // rh2  = hs @ w3 + b3
  k_router_mask<<<TOK / 8, 256, 0, stream>>>(buf0, rw_out, r_out_b, maskp, TOK, RH);

  // ---- experts (dense, masked mean accumulation) ----
  const float inv_e = 1.0f / (float)NE;
  for (int e = 0; e < NE; ++e) {
    cvt(e_in_w  + (size_t)e * DIMD * HH, ew_in,  (size_t)DIMD * HH);
    cvt(es_w1   + (size_t)e * HH * HH,   ew1,    (size_t)HH * HH);
    cvt(es_w2   + (size_t)e * HH * HH,   ew2,    (size_t)HH * HH);
    cvt(es_w3   + (size_t)e * HH * HH,   ew3,    (size_t)HH * HH);
    cvt(e_out_w + (size_t)e * HH * DIMD, ew_out, (size_t)HH * DIMD);

    gemm0(x_bf, ew_in, e_in_b + (size_t)e * HH, buf0, TOK, HH, DIMD);      // h
    gemm0(buf0, ew2,   es_b2  + (size_t)e * HH, buf1, TOK, HH, HH);        // h2
    gemm1(buf0, ew1,   es_b1  + (size_t)e * HH, buf1, buf1, TOK, HH, HH);  // hs
    gemm0(buf1, ew3,   es_b3  + (size_t)e * HH, buf0, TOK, HH, HH);        // g
    gemm2(buf0, ew_out, e_out_b + (size_t)e * DIMD, e, inv_e, out,
          TOK, DIMD, HH);                                                  // out += masked
  }
}
